// ParikhRankModel_61572651155771
// MI455X (gfx1250) — matrix operand order repaired and checked by reading, not compiled, not run on hardware
//
#include <hip/hip_runtime.h>
#include <hip/hip_bf16.h>

// ---------------------------------------------------------------------------
// ParikhRankModel fused forward for gfx1250 (MI455X).
// bf16 WMMA (v_wmma_f32_16x16x32_bf16), fp32 accumulate, TDM staging,
// 128-bit fragment loads everywhere (weights pre-transposed to N-major).
// ---------------------------------------------------------------------------

#define BB 32
#define CC 64
#define SS 128
#define ZZ 64
#define DD 256
#define GG 256

typedef __attribute__((ext_vector_type(16))) __bf16 v16bf;
typedef __attribute__((ext_vector_type(8)))  float  v8f;
typedef __attribute__((ext_vector_type(4)))  unsigned int u32x4;
typedef __attribute__((ext_vector_type(8)))  int          i32x8;
typedef __attribute__((ext_vector_type(4)))  int          i32x4;

struct bf16x16_bits { uint4 lo, hi; };  // 32 B, same size as v16bf
__device__ __forceinline__ v16bf cast16(uint4 lo, uint4 hi) {
  bf16x16_bits t{lo, hi};
  return __builtin_bit_cast(v16bf, t);
}
struct bf16x8 { __bf16 h[8]; };         // 16 B, same size as uint4
__device__ __forceinline__ uint4 pack8_bf16(float4 f0, float4 f1) {
  bf16x8 t = {{(__bf16)f0.x, (__bf16)f0.y, (__bf16)f0.z, (__bf16)f0.w,
               (__bf16)f1.x, (__bf16)f1.y, (__bf16)f1.z, (__bf16)f1.w}};
  return __builtin_bit_cast(uint4, t);
}

// ---- WMMA fragment helpers (contiguous A/B k order, C/D per CDNA5 ISA 7.12.2, wave32) ----------

// A matrix 16x32 bf16, row-major source. Lane L<16: row M=L, K {0..15};
// lane>=16: row M=L-16, K {16..31}.  Elements h=0..15 are contiguous
// (k=16*(lane/16)+h) -> two 16B loads.
__device__ __forceinline__ v16bf frag_a(const __bf16* base, int ld, int lane) {
  const __bf16* p = base + (lane & 15) * ld + ((lane >> 4) << 4);
  return cast16(*(const uint4*)(p), *(const uint4*)(p + 8));
}

// B matrix 32x16 bf16 from an N-major source: element (k,n) at base[n*ld+k].
// Lane: n = lane%16, k = 16*(lane/16) + h -> 16 contiguous elements.
__device__ __forceinline__ v16bf frag_b_nmaj(const __bf16* base, int ld, int lane) {
  const __bf16* p = base + (lane & 15) * ld + ((lane >> 4) << 4);
  return cast16(*(const uint4*)(p), *(const uint4*)(p + 8));
}

// B matrix 32x16 bf16 from a K-major source (strided; only used for the small
// alpha_j / beta_i contractions out of LDS).
__device__ __forceinline__ v16bf frag_b_k(const __bf16* base, int ld, int lane) {
  v16bf b;
  const __bf16* p = base + ((lane >> 4) << 4) * ld + (lane & 15);
#pragma unroll
  for (int h = 0; h < 16; ++h) b[h] = p[h * ld];
  return b;
}

__device__ __forceinline__ v8f wmma_bf16(v16bf a, v16bf b, v8f c) {
  return __builtin_amdgcn_wmma_f32_16x16x32_bf16(
      false, a, false, b, (short)0, c, false, false);
}

// C/D 16x16 f32: vgpr r -> M = r + 8*(lane/16), N = lane%16.
__device__ __forceinline__ void store_d_relu_bf16(__bf16* dst, int ld, int lane, v8f c) {
  const int n = lane & 15, mb = (lane >> 4) << 3;
#pragma unroll
  for (int r = 0; r < 8; ++r) dst[(mb + r) * ld + n] = (__bf16)fmaxf(c[r], 0.0f);
}
__device__ __forceinline__ void store_d_bf16(__bf16* dst, int ld, int lane, v8f c) {
  const int n = lane & 15, mb = (lane >> 4) << 3;
#pragma unroll
  for (int r = 0; r < 8; ++r) dst[(mb + r) * ld + n] = (__bf16)c[r];
}
__device__ __forceinline__ void store_d_f32(float* dst, int ld, int lane, v8f c) {
  const int n = lane & 15, mb = (lane >> 4) << 3;
#pragma unroll
  for (int r = 0; r < 8; ++r) dst[(mb + r) * ld + n] = c[r];
}
__device__ __forceinline__ void accum_cols_relu(float* sV, int n0, int lane, v8f c) {
  float p = 0.0f;
#pragma unroll
  for (int r = 0; r < 8; ++r) p += fmaxf(c[r], 0.0f);
  atomicAdd(&sV[n0 + (lane & 15)], p);  // ds_add_f32
}
__device__ __forceinline__ v8f splat8(float x) {
  v8f c;
#pragma unroll
  for (int r = 0; r < 8; ++r) c[r] = x;
  return c;
}

// ---- Tensor Data Mover: 1-D bf16 slab copy global -> LDS -------------------
// D# per ISA 08_async_tensor 8.3/8.4: group0 {flags, lds_addr, global_addr,
// type=2}; group1 {data_size=8B, tensor_dim0=tile_dim0=n_qwords (1-D tile)}.
// 6-arg builtin variant on this toolchain: (g0, g1, g2, g3, g_extra, cpol).
__device__ __forceinline__ void tdm_copy_1d(const void* gsrc, unsigned lds_off,
                                            unsigned n_qwords) {
  const unsigned long long ga = (unsigned long long)(uintptr_t)gsrc;
  u32x4 g0;
  g0[0] = 1u;                                            // count=1
  g0[1] = lds_off;                                       // lds_addr
  g0[2] = (unsigned)ga;                                  // global_addr[31:0]
  g0[3] = (unsigned)((ga >> 32) & 0x01FFFFFFu) | (2u << 30); // [56:32] | type=2
  i32x8 g1;
  g1[0] = (int)(3u << 16);                               // data_size=3 (8B)
  g1[1] = (int)((n_qwords & 0xFFFFu) << 16);             // tensor_dim0[15:0]
  g1[2] = (int)((n_qwords >> 16) & 0xFFFFu);             // tensor_dim0[31:16]
  g1[3] = (int)((n_qwords & 0xFFFFu) << 16);             // tile_dim0
  g1[4] = 0;                                             // tile_dim1/2 = 0
  g1[5] = (int)n_qwords;                                 // dim0_stride lo
  g1[6] = 0;
  g1[7] = 0;
  const i32x4 z4 = {0, 0, 0, 0};
  const i32x8 z8 = {0, 0, 0, 0, 0, 0, 0, 0};
  __builtin_amdgcn_tensor_load_to_lds(g0, g1, z4, z4, z8, 0);
}

// ---------------------------------------------------------------------------
// Kernel 0: FW^T (256x256) and GW^T (256x512) -> bf16, N-major, in workspace.
// ---------------------------------------------------------------------------
__global__ void convert_weights_kernel(const float* __restrict__ FW,
                                       const float* __restrict__ GW,
                                       __bf16* __restrict__ FWt,
                                       __bf16* __restrict__ GWt) {
  const int i = blockIdx.x * 256 + threadIdx.x;
  if (i < DD * DD) {
    const int n = i >> 8, k = i & 255;
    FWt[i] = (__bf16)FW[k * DD + n];
  }
  if (i < 2 * DD * GG) {
    const int g = i >> 9, k = i & 511;
    GWt[i] = (__bf16)GW[k * GG + g];
  }
}

// ---------------------------------------------------------------------------
// Kernel 1: a = EmbA[inp] (bf16) and a_m = relu(a @ FW + Fb) (bf16).
// 16 token rows per block, 8 wave32s, WMMA bf16 K=256.
// ---------------------------------------------------------------------------
__global__ __launch_bounds__(256) void parikh_premap_kernel(
    const int* __restrict__ inp, const float* __restrict__ EmbA,
    const __bf16* __restrict__ FWt, const float* __restrict__ Fb,
    __bf16* __restrict__ aEb, __bf16* __restrict__ aMb) {
  __shared__ __bf16 sA[16 * DD];
  const int tid = threadIdx.x, lane = tid & 31, wave = tid >> 5;
  const int r0 = blockIdx.x * 16;  // row into (B*S) = 4096

  // gather 16 embedding rows, 8 elements per thread iteration (vectorized)
  for (int ch = tid; ch < (16 * DD) / 8; ch += 256) {
    const int r = ch >> 5, d0 = (ch & 31) << 3;
    const float* src = EmbA + (size_t)inp[r0 + r] * DD + d0;
    const uint4 u = pack8_bf16(*(const float4*)(src), *(const float4*)(src + 4));
    *(uint4*)(sA + r * DD + d0) = u;
    *(uint4*)(aEb + (size_t)(r0 + r) * DD + d0) = u;
  }
  __syncthreads();

  for (int t = wave; t < 16; t += 8) {       // 16 N-tiles of 16
    const int n0 = t << 4;
    v8f acc = splat8(Fb[n0 + (lane & 15)]);
#pragma unroll
    for (int kk = 0; kk < 8; ++kk) {         // K = 256
      const v16bf af = frag_a(sA + kk * 32, DD, lane);
      const v16bf bf = frag_b_nmaj(FWt + n0 * DD + kk * 32, DD, lane);
      acc = wmma_bf16(af, bf, acc);
    }
    const int n = lane & 15, mb = (lane >> 4) << 3;
#pragma unroll
    for (int r = 0; r < 8; ++r)
      aMb[(size_t)(r0 + mb + r) * DD + n0 + n] = (__bf16)fmaxf(acc[r], 0.0f);
  }
}

// ---------------------------------------------------------------------------
// Kernel 2: one block per (b,c). Fully fused attention + G layers + H head.
// Dynamic LDS layout (bytes):
//   [0      ) sA      a bf16        128x256  65536
//   [65536  ) sAm     a_m bf16      128x256  65536   (reused as beta_i)
//   [131072 ) sB      bb bf16        64x256  32768
//   [163840 ) sBm     b_m bf16       64x256  32768   (reused as alpha_j)
//   [196608 ) sE      e f32         128x64   32768
//   [229376 ) sAl     alpha^T bf16   64x128  16384
//   [245760 ) sBe     beta bf16     128x64   16384
//   [262144 ) sV      v1|v2 f32     512       2048
// total 264192 B  (<= 320KB WGP LDS)
// ---------------------------------------------------------------------------
#define SMEM2_BYTES (264192)

__global__ __launch_bounds__(256) void parikh_attn_kernel(
    const int* __restrict__ cand, const float* __restrict__ EmbB,
    const __bf16* __restrict__ FWt, const float* __restrict__ Fb,
    const __bf16* __restrict__ GWt, const float* __restrict__ Gb,
    const float* __restrict__ HW, const float* __restrict__ Hb,
    const __bf16* __restrict__ aEb, const __bf16* __restrict__ aMb,
    float* __restrict__ out) {
  extern __shared__ char smem[];
  __bf16* sA  = (__bf16*)(smem + 0);
  __bf16* sAm = (__bf16*)(smem + 65536);
  __bf16* sB  = (__bf16*)(smem + 131072);
  __bf16* sBm = (__bf16*)(smem + 163840);
  float*  sE  = (float*)(smem + 196608);
  __bf16* sAl = (__bf16*)(smem + 229376);
  __bf16* sBe = (__bf16*)(smem + 245760);
  float*  sV  = (float*)(smem + 262144);
  __bf16* sBetaI = sAm;   // aliases: a_m dead after e
  __bf16* sAlJ   = sBm;   //          b_m dead after e

  const int tid = threadIdx.x, lane = tid & 31, wave = tid >> 5;
  const int bc = blockIdx.x, b = bc >> 6, c = bc & (CC - 1);

  sV[tid] = 0.0f;
  sV[tid + 256] = 0.0f;

  // ---- TDM: stage a / a_m (64KB each, bf16) into LDS via the DMA engine ----
  if (tid < 32) {
    tdm_copy_1d(aEb + (size_t)b * SS * DD, (unsigned)(uintptr_t)sA,
                (SS * DD * 2) / 8);
    tdm_copy_1d(aMb + (size_t)b * SS * DD, (unsigned)(uintptr_t)sAm,
                (SS * DD * 2) / 8);
    __builtin_amdgcn_s_wait_tensorcnt(0);
  }
  // ---- concurrently: gather bb = EmbB[cand[b,c,:]] -> bf16 (vectorized) ----
  {
    const int* cz = cand + ((size_t)b * CC + c) * ZZ;
    for (int ch = tid; ch < (ZZ * DD) / 8; ch += 256) {
      const int z = ch >> 5, d0 = (ch & 31) << 3;
      const float* src = EmbB + (size_t)cz[z] * DD + d0;
      *(uint4*)(sB + z * DD + d0) =
          pack8_bf16(*(const float4*)(src), *(const float4*)(src + 4));
    }
  }
  __syncthreads();

  // ---- Phase B: b_m = relu(bb @ FW + Fb)   (64x256, K=256) ----
  for (int t = wave; t < 64; t += 8) {
    const int z0 = (t >> 4) << 4, n0 = (t & 15) << 4;
    v8f acc = splat8(Fb[n0 + (lane & 15)]);
#pragma unroll
    for (int kk = 0; kk < 8; ++kk) {
      const v16bf af = frag_a(sB + z0 * DD + kk * 32, DD, lane);
      const v16bf bf = frag_b_nmaj(FWt + n0 * DD + kk * 32, DD, lane);
      acc = wmma_bf16(af, bf, acc);
    }
    store_d_relu_bf16(sBm + z0 * DD + n0, DD, lane, acc);
  }
  __syncthreads();

  // ---- Phase C: e = a_m @ b_m^T   (128x64, K=256); b_m is N-major for B ----
  for (int t = wave; t < 32; t += 8) {
    const int s0 = (t >> 2) << 4, z0 = (t & 3) << 4;
    v8f acc = splat8(0.0f);
#pragma unroll
    for (int kk = 0; kk < 8; ++kk) {
      const v16bf af = frag_a(sAm + s0 * DD + kk * 32, DD, lane);
      const v16bf bf = frag_b_nmaj(sBm + z0 * DD + kk * 32, DD, lane);
      acc = wmma_bf16(af, bf, acc);
    }
    store_d_f32(sE + s0 * ZZ + z0, ZZ, lane, acc);
  }
  __syncthreads();

  // ---- Phase D: dual softmax on e ----
  if (tid < ZZ) {  // alpha over s (columns); stored transposed [z][s]
    const int z = tid;
    float mx = -3.4e38f;
    for (int s = 0; s < SS; ++s) mx = fmaxf(mx, sE[s * ZZ + z]);
    float sum = 0.0f;
    for (int s = 0; s < SS; ++s) sum += __expf(sE[s * ZZ + z] - mx);
    const float inv = 1.0f / sum;
    for (int s = 0; s < SS; ++s)
      sAl[z * SS + s] = (__bf16)(__expf(sE[s * ZZ + z] - mx) * inv);
  }
  if (tid >= 128) {  // beta over z (rows); stored [s][z]
    const int s = tid - 128;
    float mx = -3.4e38f;
    for (int z = 0; z < ZZ; ++z) mx = fmaxf(mx, sE[s * ZZ + z]);
    float sum = 0.0f;
    for (int z = 0; z < ZZ; ++z) sum += __expf(sE[s * ZZ + z] - mx);
    const float inv = 1.0f / sum;
    for (int z = 0; z < ZZ; ++z)
      sBe[s * ZZ + z] = (__bf16)(__expf(sE[s * ZZ + z] - mx) * inv);
  }
  __syncthreads();

  // ---- Phase E: alpha_j = alpha^T @ a (64x256, K=128);
  //               beta_i  = beta @ bb  (128x256, K=64) ----
  for (int t = wave; t < 64; t += 8) {
    const int z0 = (t >> 4) << 4, d0 = (t & 15) << 4;
    v8f acc = splat8(0.0f);
#pragma unroll
    for (int kk = 0; kk < 4; ++kk) {
      const v16bf af = frag_a(sAl + z0 * SS + kk * 32, SS, lane);
      const v16bf bf = frag_b_k(sA + (kk * 32) * DD + d0, DD, lane);
      acc = wmma_bf16(af, bf, acc);
    }
    store_d_bf16(sAlJ + z0 * DD + d0, DD, lane, acc);
  }
  for (int t = wave; t < 128; t += 8) {
    const int s0 = (t >> 4) << 4, d0 = (t & 15) << 4;
    v8f acc = splat8(0.0f);
#pragma unroll
    for (int kk = 0; kk < 2; ++kk) {
      const v16bf af = frag_a(sBe + s0 * ZZ + kk * 32, ZZ, lane);
      const v16bf bf = frag_b_k(sB + (kk * 32) * DD + d0, DD, lane);
      acc = wmma_bf16(af, bf, acc);
    }
    store_d_bf16(sBetaI + s0 * DD + d0, DD, lane, acc);
  }
  __syncthreads();

  // ---- Phase F: v_1 = sum_s relu([a|beta_i] @ GW + Gb)  (K=512) ----
  for (int t = wave; t < 128; t += 8) {
    const int s0 = (t >> 4) << 4, g0 = (t & 15) << 4;
    v8f acc = splat8(Gb[g0 + (lane & 15)]);
#pragma unroll
    for (int kk = 0; kk < 8; ++kk) {
      const v16bf af = frag_a(sA + s0 * DD + kk * 32, DD, lane);
      const v16bf bf = frag_b_nmaj(GWt + g0 * (2 * DD) + kk * 32, 2 * DD, lane);
      acc = wmma_bf16(af, bf, acc);
    }
#pragma unroll
    for (int kk = 0; kk < 8; ++kk) {
      const v16bf af = frag_a(sBetaI + s0 * DD + kk * 32, DD, lane);
      const v16bf bf = frag_b_nmaj(GWt + g0 * (2 * DD) + DD + kk * 32, 2 * DD, lane);
      acc = wmma_bf16(af, bf, acc);
    }
    accum_cols_relu(sV, g0, lane, acc);
  }
  // ---- Phase G: v_2 = sum_z relu([bb|alpha_j] @ GW + Gb) ----
  for (int t = wave; t < 64; t += 8) {
    const int z0 = (t >> 4) << 4, g0 = (t & 15) << 4;
    v8f acc = splat8(Gb[g0 + (lane & 15)]);
#pragma unroll
    for (int kk = 0; kk < 8; ++kk) {
      const v16bf af = frag_a(sB + z0 * DD + kk * 32, DD, lane);
      const v16bf bf = frag_b_nmaj(GWt + g0 * (2 * DD) + kk * 32, 2 * DD, lane);
      acc = wmma_bf16(af, bf, acc);
    }
#pragma unroll
    for (int kk = 0; kk < 8; ++kk) {
      const v16bf af = frag_a(sAlJ + z0 * DD + kk * 32, DD, lane);
      const v16bf bf = frag_b_nmaj(GWt + g0 * (2 * DD) + DD + kk * 32, 2 * DD, lane);
      acc = wmma_bf16(af, bf, acc);
    }
    accum_cols_relu(sV + GG, g0, lane, acc);
  }
  __syncthreads();

  // ---- Phase H: y = [v1|v2] . HW + Hb  (wave 0) ----
  if (wave == 0) {
    float p = 0.0f;
    for (int i = lane; i < 2 * GG; i += 32) p += sV[i] * HW[i];
#pragma unroll
    for (int off = 16; off > 0; off >>= 1) p += __shfl_xor(p, off, 32);
    if (lane == 0) out[bc] = p + Hb[0];
  }
}

// ---------------------------------------------------------------------------
// Host launcher
// ---------------------------------------------------------------------------
extern "C" void kernel_launch(void* const* d_in, const int* in_sizes, int n_in,
                              void* d_out, int out_size, void* d_ws, size_t ws_size,
                              hipStream_t stream) {
  const int*   inp  = (const int*)d_in[0];    // (B,S)
  const int*   cand = (const int*)d_in[1];    // (B,C,Z)
  // d_in[2] alignments, d_in[3] align_entropies: unused in forward
  const float* EmbA = (const float*)d_in[4];  // (VI,D)
  const float* EmbB = (const float*)d_in[5];  // (VO,D)
  const float* FW   = (const float*)d_in[6];  // (D,D)
  const float* Fb   = (const float*)d_in[7];  // (D)
  const float* GW   = (const float*)d_in[8];  // (2D,G)
  const float* Gb   = (const float*)d_in[9];  // (G)
  const float* HW   = (const float*)d_in[10]; // (2G,1)
  const float* Hb   = (const float*)d_in[11]; // (1)
  float* out = (float*)d_out;                 // (B,C)

  char* ws = (char*)d_ws;
  __bf16* FWt = (__bf16*)(ws);                           // 256*256*2 = 131072
  __bf16* GWt = (__bf16*)(ws + 131072);                  // 256*512*2 = 262144
  __bf16* aEb = (__bf16*)(ws + 131072 + 262144);         // 2 MiB
  __bf16* aMb = (__bf16*)(ws + 131072 + 262144 + 2097152);

  convert_weights_kernel<<<(2 * DD * GG + 255) / 256, 256, 0, stream>>>(FW, GW, FWt, GWt);
  parikh_premap_kernel<<<(BB * SS) / 16, 256, 0, stream>>>(inp, EmbA, FWt, Fb, aEb, aMb);
  parikh_attn_kernel<<<BB * CC, 256, SMEM2_BYTES, stream>>>(
      cand, EmbB, FWt, Fb, GWt, Gb, HW, Hb, aEb, aMb, out);
}
